// BatchSplitFF_79834852098395
// MI455X (gfx1250) — matrix operand order repaired and checked by reading, not compiled, not run on hardware
//
#include <hip/hip_runtime.h>

// ---------------------------------------------------------------------------
// BatchSplitFF for MI455X (gfx1250, wave32, WMMA + TDM).
//  - gating:  f32 WMMA 16x16x4 (full-precision argmax) -> selidx in d_ws
//  - FFN:     bf16 WMMA 16x16x32, f32 accumulate (memory-roofline path)
//  - weights: one-time bf16 transpose into d_ws, then staged per-tile by the
//             Tensor Data Mover (tensor_load_to_lds + s_wait_tensorcnt)
//  - unpermute: global_atomic_add_f32 into zero-initialized d_out
// ---------------------------------------------------------------------------

typedef __attribute__((ext_vector_type(2)))  float        v2f;
typedef __attribute__((ext_vector_type(8)))  float        v8f;
typedef __attribute__((ext_vector_type(16))) __bf16       v16bf;
typedef __attribute__((ext_vector_type(4)))  unsigned int u32x4;
typedef __attribute__((ext_vector_type(8)))  int          i32x8;
typedef __attribute__((ext_vector_type(4)))  int          i32x4;

#if defined(__has_builtin)
#if __has_builtin(__builtin_amdgcn_wmma_f32_16x16x4_f32)
#define GATE_WMMA 1
#endif
#if __has_builtin(__builtin_amdgcn_tensor_load_to_lds) && \
    __has_builtin(__builtin_amdgcn_s_wait_tensorcnt)
#define HAVE_TDM 1
#endif
#endif
#ifndef GATE_WMMA
#define GATE_WMMA 0
#endif
#ifndef HAVE_TDM
#define HAVE_TDM 0
#endif

#define NGROUP 1024   // 8*2048/16
#define DM     1024
#define NSLOT  64     // 16 experts * 4 sets
#define ESZ    64
#define STR    72     // bf16 LDS row stride (144B = 16B-aligned, bank-skewed)

// workspace layout (bytes)
#define WS_SELI  0u
#define WS_F1T   (1u << 20)             // 4096 x 1024 bf16 = 8 MB  [j*64+f][d]
#define WS_F2T   (10u << 20)            // 64 x 1024 x 64 bf16 = 8 MB [j][d][f]
#define WS_NEED  ((size_t)(18u << 20))

// f32 -> bf16 RNE via native conversion
static __device__ __forceinline__ unsigned short b16(float f) {
    union { __bf16 h; unsigned short s; } u;
    u.h = (__bf16)f;
    return u.s;
}
static __device__ __forceinline__ unsigned int pk2(float a, float b) {
    return (unsigned int)b16(a) | ((unsigned int)b16(b) << 16);
}

// 16-bf16 WMMA fragment = two b128 LDS reads (p0/p1 16B-aligned)
static __device__ __forceinline__ v16bf ldfrag(const unsigned short* p0,
                                               const unsigned short* p1) {
    union { uint4 u[2]; v16bf v; } t;
    t.u[0] = *(const uint4*)p0;
    t.u[1] = *(const uint4*)p1;
    return t.v;
}

static __device__ __forceinline__ void atomic_add_f32(float* p, float v) {
    asm volatile("global_atomic_add_f32 %0, %1, off" :: "v"(p), "v"(v) : "memory");
}

#if HAVE_TDM
// Issue one 2-D TDM tile load (bf16 data): tile t0 x t1 from a row-major
// tensor (row stride s0 elements) into LDS at lds_off, padding each
// 128B (=64 bf16) row with 16B -> LDS row stride 144B (= STR bf16).
static __device__ __forceinline__ void tdm_load_2d(unsigned lds_off, const void* gp,
                                                   unsigned td0, unsigned td1,
                                                   unsigned t0, unsigned t1,
                                                   unsigned s0) {
    unsigned long long ga = (unsigned long long)(size_t)gp;
    u32x4 g0;
    g0.x = 1u;                                        // count=1, user descriptor
    g0.y = lds_off;                                   // LDS byte address
    g0.z = (unsigned)ga;                              // global_addr[31:0]
    g0.w = (unsigned)((ga >> 32) & 0x01FFFFFFull)     // global_addr[56:32]
           | 0x80000000u;                             // type=2 ("image")
    i32x8 g1;
    g1[0] = (int)((1u << 16)      // data_size = 2B
                | (1u << 20)      // pad_enable
                | (4u << 22)      // pad_interval: 32 DWORDs (=128B)
                | (3u << 25));    // pad_amount:   4 DWORDs (=16B)
    g1[1] = (int)((td0 & 0xFFFFu) << 16);                              // tensor_dim0 lo
    g1[2] = (int)(((td0 >> 16) & 0xFFFFu) | ((td1 & 0xFFFFu) << 16));  // dim0 hi | dim1 lo
    g1[3] = (int)(((td1 >> 16) & 0xFFFFu) | (t0 << 16));               // dim1 hi | tile0
    g1[4] = (int)(t1);                                                 // tile1 | tile2=0
    g1[5] = (int)s0;                                                   // dim0_stride lo
    g1[6] = 0;
    g1[7] = 0;
    i32x4 zz = {0, 0, 0, 0};
#if __clang_major__ >= 23
    i32x8 z8 = {0, 0, 0, 0, 0, 0, 0, 0};
    __builtin_amdgcn_tensor_load_to_lds(g0, g1, zz, zz, z8, 0);
#else
    __builtin_amdgcn_tensor_load_to_lds(g0, g1, zz, zz, 0);
#endif
}
#endif

// ---------------------------------------------------------------------------
__global__ __launch_bounds__(256) void zero_kernel(float4* __restrict__ out, int n4) {
    int i = blockIdx.x * 256 + threadIdx.x;
    if (i < n4) out[i] = float4{0.f, 0.f, 0.f, 0.f};
}

// ---------------------------------------------------------------------------
// Gating: one block per group; wave w owns slot tile w*16..w*16+15.
// ---------------------------------------------------------------------------
__global__ __launch_bounds__(128) void gate_kernel(const float* __restrict__ x,
                                                   const float* __restrict__ ctl,
                                                   int* __restrict__ selidx) {
    const int g   = blockIdx.x;
    const int tid = threadIdx.x;

    __shared__ float At[16 * 66];
    __shared__ float Ct[64 * 66];

#if GATE_WMMA
    v8f acc = {};
    const int lane  = tid & 31;
    const int w     = tid >> 5;
    const int laneN = lane & 15;
    const int hk    = (lane >> 4) * 2;
    const int arow  = laneN * 66;
    const int jrow  = (w * 16 + laneN) * 66;
#else
    float accs[16];
    if (tid < 64) {
        #pragma unroll
        for (int t = 0; t < 16; ++t) accs[t] = 0.f;
    }
#endif

    for (int kc = 0; kc < 16; ++kc) {
        #pragma unroll
        for (int i = 0; i < 8; ++i) {
            int idx = i * 128 + tid, r = idx >> 6, c = idx & 63;
            At[r * 66 + c] = x[(size_t)(g * 16 + r) * DM + kc * 64 + c];
        }
        #pragma unroll
        for (int i = 0; i < 32; ++i) {
            int idx = i * 128 + tid, d = idx >> 6, j = idx & 63;
            Ct[j * 66 + d] = ctl[(size_t)(kc * 64 + d) * NSLOT + j];
        }
        __syncthreads();

#if GATE_WMMA
        #pragma unroll
        for (int ks = 0; ks < 16; ++ks) {
            int k0 = ks * 4;
            v2f a = *(const v2f*)(At + arow + k0 + hk);
            v2f b = *(const v2f*)(Ct + jrow + k0 + hk);
            acc = __builtin_amdgcn_wmma_f32_16x16x4_f32(false, a, false, b,
                                                        (short)0, acc, false, false);
        }
#else
        if (tid < 64) {
            for (int d = 0; d < 64; ++d) {
                float c = Ct[tid * 66 + d];
                #pragma unroll
                for (int t = 0; t < 16; ++t) accs[t] += At[t * 66 + d] * c;
            }
        }
#endif
        __syncthreads();
    }

    const float tiestep = 1e-6f / 15.f;
#if GATE_WMMA
    int   mbase = (lane >> 4) * 8;
    float bestv = acc[0] + (float)mbase * tiestep;
    int   besti = mbase;
    #pragma unroll
    for (int r = 1; r < 8; ++r) {
        float v = acc[r] + (float)(mbase + r) * tiestep;
        if (v >= bestv) { bestv = v; besti = mbase + r; }
    }
    float ov = __shfl(bestv, laneN + 16, 32);
    int   oi = __shfl(besti, laneN + 16, 32);
    if (lane < 16) {
        int sel = (ov >= bestv) ? oi : besti;
        selidx[g * NSLOT + w * 16 + lane] = sel;
    }
#else
    if (tid < 64) {
        float bestv = accs[0]; int besti = 0;
        #pragma unroll
        for (int t = 1; t < 16; ++t) {
            float v = accs[t] + (float)t * tiestep;
            if (v >= bestv) { bestv = v; besti = t; }
        }
        selidx[g * NSLOT + tid] = besti;
    }
#endif
}

#if HAVE_TDM
// ---------------------------------------------------------------------------
// Weight prep: f1T[j*64+f][d] = bf16(f1[d][j*64+f]); f2T[j][d][f] = bf16(f2[j][f][d])
// LDS-tiled transposes, coalesced on both sides.
// ---------------------------------------------------------------------------
__global__ __launch_bounds__(256) void prep_f1(const float* __restrict__ f1,
                                               unsigned short* __restrict__ f1T) {
    __shared__ float T[64 * 65];
    const int jf0 = blockIdx.x * 64;        // 64 tiles over 4096 (j,f) columns
    const int d0  = blockIdx.y * 64;        // 16 tiles over 1024 d rows
    const int tid = threadIdx.x;
    #pragma unroll
    for (int i = 0; i < 16; ++i) {
        int idx = i * 256 + tid, r = idx >> 6, c = idx & 63;
        T[r * 65 + c] = f1[(size_t)(d0 + r) * 4096 + jf0 + c];
    }
    __syncthreads();
    #pragma unroll
    for (int i = 0; i < 16; ++i) {
        int idx = i * 256 + tid, c = idx >> 6, r = idx & 63;
        f1T[(size_t)(jf0 + c) * 1024 + d0 + r] = b16(T[r * 65 + c]);
    }
}

__global__ __launch_bounds__(256) void prep_f2(const float* __restrict__ f2,
                                               unsigned short* __restrict__ f2T) {
    __shared__ float T[64 * 65];
    const int j  = blockIdx.x;              // 64 slots
    const int d0 = blockIdx.y * 64;         // 16 tiles over 1024 d
    const int tid = threadIdx.x;
    #pragma unroll
    for (int i = 0; i < 16; ++i) {
        int idx = i * 256 + tid, f = idx >> 6, dl = idx & 63;
        T[f * 65 + dl] = f2[(size_t)j * 65536 + (size_t)f * 1024 + d0 + dl];
    }
    __syncthreads();
    #pragma unroll
    for (int i = 0; i < 16; ++i) {
        int idx = i * 256 + tid, dl = idx >> 6, f = idx & 63;
        f2T[(size_t)j * 65536 + (size_t)(d0 + dl) * 64 + f] = b16(T[f * 65 + dl]);
    }
}
#endif  // HAVE_TDM

// ---------------------------------------------------------------------------
// Shared FFN body. TDM variant stages B operands with tensor_load_to_lds from
// pre-transposed bf16 weights; fallback stages+transposes f32 weights in VALU.
// Block = 256 threads = 8 waves; wave w owns M strip w*16..w*16+15.
// ---------------------------------------------------------------------------
#define FFN_PROLOGUE()                                                          \
    const int j    = blockIdx.x;                                                \
    const int gt   = blockIdx.y;                                                \
    const int tid  = threadIdx.x;                                               \
    const int lane = tid & 31, wave = tid >> 5;                                 \
    const int laneN = lane & 15, half = lane >> 4;                              \
    __shared__ unsigned short regA[128 * STR];                                  \
    __shared__ unsigned short regB[128 * STR];                                  \
    __shared__ int tokRow[128];                                                 \
    if (tid < 128) {                                                            \
        int g = gt * 128 + tid;                                                 \
        tokRow[tid] = g * 16 + selidx[g * NSLOT + j];                           \
    }                                                                           \
    __syncthreads();                                                            \
    float biasv[4];                                                             \
    _Pragma("unroll")                                                           \
    for (int nt = 0; nt < 4; ++nt) biasv[nt] = bias[j * ESZ + nt * 16 + laneN]; \
    v8f acc[4] = {};

#define FFN_STAGE_A(kc)                                                         \
    {   int rr = tid >> 4, c0 = (tid & 15) * 4;                                 \
        _Pragma("unroll")                                                       \
        for (int i = 0; i < 8; ++i) {                                           \
            int m = i * 16 + rr;                                                \
            const float* src = x + (size_t)tokRow[m] * DM + (kc) * 64 + c0;     \
            const float4 v = *(const float4*)src;                               \
            if ((kc) < 15) __builtin_prefetch(src + 64, 0, 1);                  \
            uint2 p; p.x = pk2(v.x, v.y); p.y = pk2(v.z, v.w);                  \
            *(uint2*)(&regA[m * STR + c0]) = p;                                 \
        }                                                                       \
    }

#define FFN_GEMM1_STEP()                                                        \
    {   const int arow = (wave * 16 + laneN) * STR;                             \
        _Pragma("unroll")                                                       \
        for (int kk = 0; kk < 2; ++kk) {                                        \
            int aoff = kk * 32 + half * 8;                                      \
            v16bf af = ldfrag(&regA[arow + aoff], &regA[arow + aoff + 16]);     \
            int boff = kk * 32 + half * 8;                                      \
            _Pragma("unroll")                                                   \
            for (int nt = 0; nt < 4; ++nt) {                                    \
                int brow = (nt * 16 + laneN) * STR;                             \
                v16bf bf = ldfrag(&regB[brow + boff], &regB[brow + boff + 16]); \
                acc[nt] = __builtin_amdgcn_wmma_f32_16x16x32_bf16(              \
                    false, af, false, bf, (short)0, acc[nt], false, false);     \
            }                                                                   \
        }                                                                       \
    }

#define FFN_INNER_STORE()                                                       \
    {   int irow0 = wave * 16;                                                  \
        _Pragma("unroll")                                                       \
        for (int nt = 0; nt < 4; ++nt) {                                        \
            int f = nt * 16 + laneN;                                            \
            _Pragma("unroll")                                                   \
            for (int r = 0; r < 8; ++r) {                                       \
                float v = acc[nt][r] + biasv[nt];                               \
                v = v > 0.f ? v : 0.f;                                          \
                regA[(irow0 + r + half * 8) * STR + f] = b16(v);                \
            }                                                                   \
        }                                                                       \
    }                                                                           \
    int trow[8];                                                                \
    _Pragma("unroll")                                                           \
    for (int r = 0; r < 8; ++r) trow[r] = tokRow[wave * 16 + r + half * 8];     \
    __syncthreads();

#define FFN_GEMM2_STEP(nc)                                                      \
    {   const int arow = (wave * 16 + laneN) * STR;                             \
        _Pragma("unroll")                                                       \
        for (int nt2 = 0; nt2 < 8; ++nt2) {                                     \
            v8f c = {};                                                         \
            _Pragma("unroll")                                                   \
            for (int kk = 0; kk < 2; ++kk) {                                    \
                int aoff = kk * 32 + half * 8;                                  \
                v16bf af = ldfrag(&regA[arow + aoff], &regA[arow + aoff + 16]); \
                int brow = (nt2 * 16 + laneN) * STR;                            \
                int boff = kk * 32 + half * 8;                                  \
                v16bf bf = ldfrag(&regB[brow + boff], &regB[brow + boff + 16]); \
                c = __builtin_amdgcn_wmma_f32_16x16x32_bf16(                    \
                    false, af, false, bf, (short)0, c, false, false);           \
            }                                                                   \
            int col = (nc) * 128 + nt2 * 16 + laneN;                            \
            _Pragma("unroll")                                                   \
            for (int r = 0; r < 8; ++r)                                         \
                atomic_add_f32(out + (size_t)trow[r] * DM + col, c[r]);         \
        }                                                                       \
    }

// -------- fallback FFN (pure VALU staging, f32 weights) --------
__global__ __launch_bounds__(256) void ffn_kernel(const float* __restrict__ x,
                                                  const float* __restrict__ f1,
                                                  const float* __restrict__ f2,
                                                  const float* __restrict__ bias,
                                                  const int*   __restrict__ selidx,
                                                  float*       __restrict__ out) {
    FFN_PROLOGUE();

    for (int kc = 0; kc < 16; ++kc) {
        FFN_STAGE_A(kc);
        {   int dl0 = tid >> 4, f0 = (tid & 15) * 4;
            #pragma unroll
            for (int i = 0; i < 4; ++i) {
                int dl = i * 16 + dl0;
                const float4 v = *(const float4*)(f1 + (size_t)(kc * 64 + dl) * 4096 + j * ESZ + f0);
                regB[(f0 + 0) * STR + dl] = b16(v.x);
                regB[(f0 + 1) * STR + dl] = b16(v.y);
                regB[(f0 + 2) * STR + dl] = b16(v.z);
                regB[(f0 + 3) * STR + dl] = b16(v.w);
            }
        }
        __syncthreads();
        FFN_GEMM1_STEP();
        __syncthreads();
    }

    FFN_INNER_STORE();

    for (int nc = 0; nc < 8; ++nc) {
        {   int fr0 = tid >> 5, d0 = lane * 4;
            #pragma unroll
            for (int i = 0; i < 8; ++i) {
                int fr = i * 8 + fr0;
                const float4 v = *(const float4*)(f2 + (size_t)j * 65536 +
                                                  (size_t)fr * DM + nc * 128 + d0);
                regB[(d0 + 0) * STR + fr] = b16(v.x);
                regB[(d0 + 1) * STR + fr] = b16(v.y);
                regB[(d0 + 2) * STR + fr] = b16(v.z);
                regB[(d0 + 3) * STR + fr] = b16(v.w);
            }
        }
        __syncthreads();
        FFN_GEMM2_STEP(nc);
        __syncthreads();
    }
}

#if HAVE_TDM
// -------- TDM FFN (B operands staged by the Tensor Data Mover) --------
__global__ __launch_bounds__(256) void ffn_kernel_tdm(const float* __restrict__ x,
                                                      const unsigned short* __restrict__ f1T,
                                                      const unsigned short* __restrict__ f2T,
                                                      const float* __restrict__ bias,
                                                      const int*   __restrict__ selidx,
                                                      float*       __restrict__ out) {
    FFN_PROLOGUE();
    const unsigned ldsB = (unsigned)(size_t)(void*)regB;

    for (int kc = 0; kc < 16; ++kc) {
        // f1T tile: rows f=0..63 (stride 1024 bf16), cols d = kc*64..+63
        if (wave == 0)
            tdm_load_2d(ldsB, f1T + (size_t)(j * 64) * 1024 + kc * 64,
                        /*td0=*/1024, /*td1=*/64, /*t0=*/64, /*t1=*/64, /*s0=*/1024);
        FFN_STAGE_A(kc);
        if (wave == 0) __builtin_amdgcn_s_wait_tensorcnt(0);
        __syncthreads();
        FFN_GEMM1_STEP();
        __syncthreads();
    }

    FFN_INNER_STORE();

    for (int nc = 0; nc < 8; ++nc) {
        // f2T tile: rows d = nc*128..+127 (stride 64 bf16), cols f=0..63
        if (wave == 0) {
            tdm_load_2d(ldsB, f2T + (size_t)j * 65536 + (size_t)(nc * 128) * 64,
                        /*td0=*/64, /*td1=*/1024, /*t0=*/64, /*t1=*/128, /*s0=*/64);
            __builtin_amdgcn_s_wait_tensorcnt(0);
        }
        __syncthreads();
        FFN_GEMM2_STEP(nc);
        __syncthreads();
    }
}
#endif  // HAVE_TDM

// ---------------------------------------------------------------------------
extern "C" void kernel_launch(void* const* d_in, const int* in_sizes, int n_in,
                              void* d_out, int out_size, void* d_ws, size_t ws_size,
                              hipStream_t stream) {
    (void)in_sizes; (void)n_in;
    const float* x    = (const float*)d_in[0];
    const float* ctl  = (const float*)d_in[1];
    const float* f1   = (const float*)d_in[2];
    const float* f2   = (const float*)d_in[3];
    const float* bias = (const float*)d_in[4];
    float* out  = (float*)d_out;
    int*   seli = (int*)((char*)d_ws + WS_SELI);

    int n4 = out_size / 4;
    zero_kernel<<<(n4 + 255) / 256, 256, 0, stream>>>((float4*)out, n4);
    gate_kernel<<<NGROUP, 128, 0, stream>>>(x, ctl, seli);

#if HAVE_TDM
    if (ws_size >= WS_NEED) {
        unsigned short* f1T = (unsigned short*)((char*)d_ws + WS_F1T);
        unsigned short* f2T = (unsigned short*)((char*)d_ws + WS_F2T);
        prep_f1<<<dim3(64, 16), 256, 0, stream>>>(f1, f1T);
        prep_f2<<<dim3(64, 16), 256, 0, stream>>>(f2, f2T);
        ffn_kernel_tdm<<<dim3(NSLOT, 8), 256, 0, stream>>>(x, f1T, f2T, bias, seli, out);
        return;
    }
#endif
    ffn_kernel<<<dim3(NSLOT, 8), 256, 0, stream>>>(x, f1, f2, bias, seli, out);
}
